// AdaptiveMoDBlock_53068615909663
// MI455X (gfx1250) — matrix operand order repaired and checked by reading, not compiled, not run on hardware
//
#include <hip/hip_runtime.h>
#include <cstdint>

#define B_    4
#define S_    4096
#define D_    2048
#define DC_   512
#define DFF_  8192
#define CHUNK_ 2048
#define ASTR  48   // padded LDS stride (elements); 96B rows = 64B data + 32B pad

typedef __attribute__((ext_vector_type(16))) __bf16 v16bf;
typedef __attribute__((ext_vector_type(8)))  __bf16 v8bf;
typedef __attribute__((ext_vector_type(8)))  float  v8f;

// ---- CDNA5 async global->LDS copy (ASYNCcnt) with safe fallback ----
#if defined(__has_builtin)
# if __has_builtin(__builtin_amdgcn_global_load_async_to_lds_b128) && \
     __has_builtin(__builtin_amdgcn_s_wait_asynccnt)
#  define USE_ASYNC_LDS 1
# endif
# if __has_builtin(__builtin_amdgcn_tensor_load_to_lds) && \
     __has_builtin(__builtin_amdgcn_s_wait_tensorcnt)
#  define USE_TDM 1
# endif
#endif

typedef int v4i_t __attribute__((vector_size(16)));                       // GCC-style int4
typedef __attribute__((address_space(1))) v4i_t gv4i;                     // global (AS1)
typedef __attribute__((address_space(3))) v4i_t lv4i;                     // LDS (AS3)
typedef unsigned int v4u_t __attribute__((vector_size(16)));
typedef int v8i_t __attribute__((vector_size(32)));

__device__ __forceinline__ void copy16_g2l(const void* gsrc, void* ldst) {
#ifdef USE_ASYNC_LDS
    __builtin_amdgcn_global_load_async_to_lds_b128(
        (gv4i*)(v4i_t*)(void*)gsrc, (lv4i*)(v4i_t*)ldst, 0, 0);
#else
    *(int4*)ldst = *(const int4*)gsrc;
#endif
}
__device__ __forceinline__ void wait_async_copies() {
#ifdef USE_ASYNC_LDS
    __builtin_amdgcn_s_wait_asynccnt(0);
#endif
}

__device__ __forceinline__ unsigned lds_addr_of(const void* p) {
    return (unsigned)(size_t)(__attribute__((address_space(3))) const void*)p;
}

#ifdef USE_TDM
// 2D TDM tile load: rows x (32 bf16) slice, LDS rows padded 64B data + 32B pad (ASTR)
__device__ __forceinline__ void tdm_load_2d(unsigned lds_byte_addr, const void* gaddr,
                                            unsigned tile_rows, unsigned tensor_d0,
                                            unsigned tensor_d1, unsigned row_stride_elems) {
    unsigned long long ga = (unsigned long long)(size_t)gaddr;
    v4u_t g0;
    g0[0] = 1u;                                            // count=1, user mode
    g0[1] = lds_byte_addr;                                 // lds_addr
    g0[2] = (unsigned)(ga & 0xFFFFFFFFu);                  // global_addr[31:0]
    g0[3] = (unsigned)((ga >> 32) & 0x1FFFFFFu) | (2u << 30); // addr[56:32] | type=2
    v8i_t g1;
    // data_size=1 (2B), pad_enable, pad_interval=3 (16 DW = 64B), pad_amount=7 (8 DW = 32B)
    g1[0] = (int)((1u << 16) | (1u << 20) | (3u << 22) | (7u << 25));
    g1[1] = (int)((tensor_d0 & 0xFFFFu) << 16);            // tensor_dim0[15:0] @bits63:48
    g1[2] = (int)(((tensor_d0 >> 16) & 0xFFFFu) | ((tensor_d1 & 0xFFFFu) << 16));
    g1[3] = (int)(((tensor_d1 >> 16) & 0xFFFFu) | (32u << 16));  // tile_dim0 = 32 elems
    g1[4] = (int)(tile_rows & 0xFFFFu);                    // tile_dim1
    g1[5] = (int)row_stride_elems;                         // tensor_dim0_stride[31:0]
    g1[6] = 0;
    g1[7] = 0;
    v4i_t z4 = {0, 0, 0, 0};
    v8i_t z8 = {0, 0, 0, 0, 0, 0, 0, 0};
    __builtin_amdgcn_tensor_load_to_lds(g0, g1, z4, z4, z8, 0);   // 6-arg toolchain form
}
#endif
__device__ __forceinline__ void tdm_wait_all() {
#ifdef USE_TDM
    __builtin_amdgcn_s_wait_tensorcnt(0);
#endif
}

__device__ __forceinline__ float gelu_exact(float x) {
    return 0.5f * x * (1.0f + erff(x * 0.70710678118654752440f));
}
__device__ __forceinline__ float sigmoidf_(float x) {
    return 1.0f / (1.0f + expf(-x));
}

// pack 8 f32 -> one 16B LDS store
__device__ __forceinline__ void cvt_store8(__bf16* dst, float4 f0, float4 f1) {
    v8bf pk;
    pk[0] = (__bf16)f0.x; pk[1] = (__bf16)f0.y; pk[2] = (__bf16)f0.z; pk[3] = (__bf16)f0.w;
    pk[4] = (__bf16)f1.x; pk[5] = (__bf16)f1.y; pk[6] = (__bf16)f1.z; pk[7] = (__bf16)f1.w;
    *(v8bf*)dst = pk;
}

// one A-fragment x four B-fragments -> 4 WMMAs (frag loads batched before MMA burst)
__device__ __forceinline__ void mma_step(const __bf16* laP, const __bf16* lbP,
                                         int wm, int wng, int lrow, int lsel, v8f acc[4]) {
    int ar = wm * 16 + lrow;
    v8bf a_lo = *(const v8bf*)&laP[ar * ASTR + lsel * 16];
    v8bf a_hi = *(const v8bf*)&laP[ar * ASTR + lsel * 16 + 8];
    v16bf af = __builtin_shufflevector(a_lo, a_hi,
        0,1,2,3,4,5,6,7,8,9,10,11,12,13,14,15);
    v16bf bfr[4];
#pragma unroll
    for (int sub = 0; sub < 4; ++sub)
        bfr[sub] = *(const v16bf*)&lbP[(wng * 64 + sub * 16 + lrow) * ASTR + lsel * 16];
#pragma unroll
    for (int sub = 0; sub < 4; ++sub)
        acc[sub] = __builtin_amdgcn_wmma_f32_16x16x32_bf16(false, af, false, bfr[sub],
                                                           (short)0, acc[sub], false, false);
}

// ---------------- mean pool over S: pooled[b,d] ----------------
__global__ void pool_kernel(const float* __restrict__ hidden, float* __restrict__ pooled) {
    int gid = blockIdx.x * 256 + threadIdx.x;      // covers B*D
    int b = gid / D_;
    int d = gid % D_;
    const float* p = hidden + (size_t)b * S_ * D_ + d;
    float s = 0.f;
    for (int i = 0; i < S_; ++i) s += p[(size_t)i * D_];
    pooled[gid] = s * (1.0f / (float)S_);
}

// ---------------- router logits: one wave per token ----------------
__global__ void logits_kernel(const float* __restrict__ hidden,
                              const float* __restrict__ rw,
                              const float* __restrict__ rb,
                              float* __restrict__ logits) {
    int wave = threadIdx.x >> 5;
    int lane = threadIdx.x & 31;
    int t = blockIdx.x * 8 + wave;                 // token over B*S
    const float* p = hidden + (size_t)t * D_;
    float s = 0.f;
    for (int i = lane; i < D_; i += 32) s += p[i] * rw[i];
    for (int m = 16; m >= 1; m >>= 1) s += __shfl_xor(s, m, 32);
    if (lane == 0) logits[t] = s + rb[0];
}

// ---------------- complexity head -> k (single block) ----------------
__global__ void comp_kernel(const float* __restrict__ pooled,
                            const float* __restrict__ cw1, const float* __restrict__ cb1,
                            const float* __restrict__ cw2, const float* __restrict__ cb2,
                            int* __restrict__ kval) {
    __shared__ float sp[D_];
    __shared__ float red[256];
    __shared__ float caps;
    int tid = threadIdx.x;
    if (tid == 0) caps = 0.f;
    for (int b = 0; b < B_; ++b) {
        for (int i = tid; i < D_; i += 256) sp[i] = pooled[b * D_ + i];
        __syncthreads();
        float part = 0.f;
        for (int j = tid; j < DC_; j += 256) {
            float acc = cb1[j];
            for (int dd = 0; dd < D_; ++dd) acc += sp[dd] * cw1[dd * DC_ + j];
            part += gelu_exact(acc) * cw2[j];
        }
        red[tid] = part;
        __syncthreads();
        for (int off = 128; off >= 1; off >>= 1) {
            if (tid < off) red[tid] += red[tid + off];
            __syncthreads();
        }
        if (tid == 0) caps += sigmoidf_(red[0] + cb2[0]);
        __syncthreads();
    }
    if (tid == 0) {
        float capacity = 0.25f + (caps * 0.25f) * 0.75f;   // MIN + mean(comp)*(MAX-MIN)
        int k = (int)((double)capacity * (double)S_);
        if (k > S_) k = S_;
        if (k < 0)  k = 0;
        kval[0] = k;
    }
}

// ---------------- rank-based top-k selection ----------------
__global__ void topk_kernel(const float* __restrict__ logits, const int* __restrict__ kval,
                            int* __restrict__ sel_idx, float* __restrict__ sel_w,
                            int* __restrict__ sel_count) {
    __shared__ float sl[S_];
    int b   = blockIdx.x >> 4;                     // 16 blocks per batch
    int blk = blockIdx.x & 15;
    int tid = threadIdx.x;
    for (int j = tid; j < S_; j += 256) sl[j] = logits[b * S_ + j];
    __syncthreads();
    int s = blk * 256 + tid;
    float v = sl[s];
    int rank = 0;
    for (int j = 0; j < S_; ++j) {
        float x = sl[j];
        rank += (x > v) || (x == v && j < s);
    }
    int k = kval[0];
    if (rank < k) {
        int pos = atomicAdd(&sel_count[b], 1);
        sel_idx[b * S_ + pos] = s;
        sel_w[b * S_ + pos]   = sigmoidf_(v);
    }
}

// ---------------- tiled transpose + f32->bf16: in[R,C] -> outT[C,R] ----------------
__global__ __launch_bounds__(256)
void transpose_bf16_kernel(const float* __restrict__ in, __bf16* __restrict__ outT,
                           int R, int C) {
    __shared__ float t[32][33];
    int c0 = blockIdx.x * 32, r0 = blockIdx.y * 32;
    int tx = threadIdx.x & 31, ty = threadIdx.x >> 5;   // 32 x 8
#pragma unroll
    for (int i = 0; i < 32; i += 8)
        t[ty + i][tx] = in[(size_t)(r0 + ty + i) * C + c0 + tx];
    __syncthreads();
#pragma unroll
    for (int i = 0; i < 32; i += 8)
        outT[(size_t)(c0 + ty + i) * R + r0 + tx] = (__bf16)t[tx][ty + i];
}

// ---------------- GEMM1: H = GELU(gather(X) @ W1 + b1), bf16 out ----------------
// BM=64, BN=128, BK=32; double-buffered LDS; async-to-LDS B staging; gather+cvt A staging
__global__ __launch_bounds__(256)
void ffn1_kernel(const float* __restrict__ hidden, const int* __restrict__ sel_idx,
                 const int* __restrict__ kval, const __bf16* __restrict__ w1t,
                 const float* __restrict__ b1, __bf16* __restrict__ Hbuf,
                 int batch, int row0) {
    __shared__ __bf16 la[2][64 * ASTR];
    __shared__ __bf16 lb[2][128 * ASTR];
    int k = kval[0];
    int tile_row = row0 + blockIdx.y * 64;
    if (tile_row >= k) return;
    int tid = threadIdx.x;
    int n0 = blockIdx.x * 128;
    int lane = tid & 31, wave = tid >> 5;
    int wm = wave >> 1, wng = wave & 1;
    int lrow = lane & 15, lsel = lane >> 4;

    // per-thread A-gather assignment (64 rows x 32 k, 8 f32 per thread)
    int arow = tid >> 2;
    int akk  = (tid & 3) * 8;
    int grow = tile_row + arow;
    bool avalid = grow < k;
    const float* asrc = avalid
        ? hidden + ((size_t)(batch * S_ + sel_idx[grow])) * D_ + akk : nullptr;
    __bf16* adst0 = &la[0][arow * ASTR + akk];
    __bf16* adst1 = &la[1][arow * ASTR + akk];

    int bnr0 = tid >> 2;                 // B staging rows (two reps of 256 chunks)
    int bkk0 = (tid & 3) * 8;
    int bnr1 = (tid + 256) >> 2;
    int bkk1 = ((tid + 256) & 3) * 8;

    float4 p0 = make_float4(0.f, 0.f, 0.f, 0.f), p1 = p0;

    // ---- prologue: stage K-tile 0 into buffer 0 ----
    if (avalid) { p0 = *(const float4*)(asrc + 0); p1 = *(const float4*)(asrc + 4); }
    copy16_g2l(w1t + (size_t)(n0 + bnr0) * D_ + 0 + bkk0, &lb[0][bnr0 * ASTR + bkk0]);
    copy16_g2l(w1t + (size_t)(n0 + bnr1) * D_ + 0 + bkk1, &lb[0][bnr1 * ASTR + bkk1]);
    cvt_store8(adst0, p0, p1);
    wait_async_copies();
    __syncthreads();

    v8f acc[4] = {};
    for (int kb = 0; kb < D_; kb += 64) {
        // -- stage (kb+32) into buf1, compute buf0 --
        if (avalid) { p0 = *(const float4*)(asrc + kb + 32); p1 = *(const float4*)(asrc + kb + 36); }
        copy16_g2l(w1t + (size_t)(n0 + bnr0) * D_ + kb + 32 + bkk0, &lb[1][bnr0 * ASTR + bkk0]);
        copy16_g2l(w1t + (size_t)(n0 + bnr1) * D_ + kb + 32 + bkk1, &lb[1][bnr1 * ASTR + bkk1]);
        mma_step(la[0], lb[0], wm, wng, lrow, lsel, acc);
        cvt_store8(adst1, p0, p1);
        wait_async_copies();
        __syncthreads();
        // -- stage (kb+64) into buf0 (if any), compute buf1 --
        bool more = (kb + 64) < D_;
        if (more) {
            if (avalid) { p0 = *(const float4*)(asrc + kb + 64); p1 = *(const float4*)(asrc + kb + 68); }
            copy16_g2l(w1t + (size_t)(n0 + bnr0) * D_ + kb + 64 + bkk0, &lb[0][bnr0 * ASTR + bkk0]);
            copy16_g2l(w1t + (size_t)(n0 + bnr1) * D_ + kb + 64 + bkk1, &lb[0][bnr1 * ASTR + bkk1]);
        }
        mma_step(la[1], lb[1], wm, wng, lrow, lsel, acc);
        if (more) cvt_store8(adst0, p0, p1);
        wait_async_copies();
        __syncthreads();
    }
    // epilogue: row m = vgpr + 8*(lane>=16)
    int mrow = blockIdx.y * 64 + wm * 16 + lsel * 8;
#pragma unroll
    for (int sub = 0; sub < 4; ++sub) {
        int n = n0 + wng * 64 + sub * 16 + lrow;
        float bias = b1[n];
#pragma unroll
        for (int v = 0; v < 8; ++v) {
            float val = gelu_exact(acc[sub][v] + bias);
            Hbuf[(size_t)(mrow + v) * DFF_ + n] = (__bf16)val;
        }
    }
}

// ---------------- GEMM2: out[b, idx] += w * (H @ W2 + b2) ----------------
// Staging via Tensor Data Mover (2 TDM tile loads / K-step) when available.
__global__ __launch_bounds__(256)
void ffn2_kernel(const __bf16* __restrict__ Hbuf, const __bf16* __restrict__ w2t,
                 const float* __restrict__ b2, const int* __restrict__ sel_idx,
                 const float* __restrict__ sel_w, const int* __restrict__ kval,
                 float* __restrict__ out, int batch, int row0) {
    __shared__ __bf16 la[2][64 * ASTR];
    __shared__ __bf16 lb[2][128 * ASTR];
    int k = kval[0];
    int tile_row = row0 + blockIdx.y * 64;
    if (tile_row >= k) return;
    int tid = threadIdx.x;
    int n0 = blockIdx.x * 128;
    int lane = tid & 31, wave = tid >> 5;
    int wm = wave >> 1, wng = wave & 1;
    int lrow = lane & 15, lsel = lane >> 4;

    const __bf16* Hbase = Hbuf + (size_t)(blockIdx.y * 64) * DFF_;
    const __bf16* Wbase = w2t + (size_t)n0 * DFF_;

#ifdef USE_TDM
#define FFN2_STAGE(buf, kb)                                                         \
    do { if (wave == 0) {                                                           \
        tdm_load_2d(lds_addr_of(&la[buf][0]), Hbase + (kb), 64, DFF_, CHUNK_, DFF_);\
        tdm_load_2d(lds_addr_of(&lb[buf][0]), Wbase + (kb), 128, DFF_, D_, DFF_);   \
    } } while (0)
#define FFN2_WAIT() tdm_wait_all()
#else
    int arow = tid >> 2;
    int akk  = (tid & 3) * 8;
    int bnr1 = (tid + 256) >> 2;
    int bkk1 = ((tid + 256) & 3) * 8;
#define FFN2_STAGE(buf, kb)                                                              \
    do {                                                                                 \
        copy16_g2l(Hbase + (size_t)arow * DFF_ + (kb) + akk, &la[buf][arow * ASTR + akk]); \
        copy16_g2l(Wbase + (size_t)arow * DFF_ + (kb) + akk, &lb[buf][arow * ASTR + akk]); \
        copy16_g2l(Wbase + (size_t)bnr1 * DFF_ + (kb) + bkk1, &lb[buf][bnr1 * ASTR + bkk1]);\
    } while (0)
#define FFN2_WAIT() wait_async_copies()
#endif

    FFN2_STAGE(0, 0);
    FFN2_WAIT();
    __syncthreads();

    v8f acc[4] = {};
    for (int kb = 0; kb < DFF_; kb += 64) {
        FFN2_STAGE(1, kb + 32);
        mma_step(la[0], lb[0], wm, wng, lrow, lsel, acc);
        FFN2_WAIT();
        __syncthreads();
        bool more = (kb + 64) < DFF_;
        if (more) FFN2_STAGE(0, kb + 64);
        mma_step(la[1], lb[1], wm, wng, lrow, lsel, acc);
        FFN2_WAIT();
        __syncthreads();
    }
#undef FFN2_STAGE
#undef FFN2_WAIT

    // epilogue: weighted scatter-add onto residual copy
    int mbase = blockIdx.y * 64 + wm * 16 + lsel * 8;
    int   idxs[8];
    float wts[8];
    bool  val[8];
#pragma unroll
    for (int v = 0; v < 8; ++v) {
        int grow = row0 + mbase + v;
        val[v] = grow < k;
        idxs[v] = val[v] ? sel_idx[grow] : 0;
        wts[v]  = val[v] ? sel_w[grow]   : 0.f;
    }
#pragma unroll
    for (int sub = 0; sub < 4; ++sub) {
        int n = n0 + wng * 64 + sub * 16 + lrow;
        float bias = b2[n];
#pragma unroll
        for (int v = 0; v < 8; ++v) {
            if (val[v]) {
                size_t o = ((size_t)(batch * S_ + idxs[v])) * D_ + n;
                out[o] += wts[v] * (acc[sub][v] + bias);
            }
        }
    }
}

extern "C" void kernel_launch(void* const* d_in, const int* in_sizes, int n_in,
                              void* d_out, int out_size, void* d_ws, size_t ws_size,
                              hipStream_t stream) {
    const float* hidden = (const float*)d_in[0];
    const float* rw  = (const float*)d_in[1];
    const float* rb  = (const float*)d_in[2];
    const float* cw1 = (const float*)d_in[3];
    const float* cb1 = (const float*)d_in[4];
    const float* cw2 = (const float*)d_in[5];
    const float* cb2 = (const float*)d_in[6];
    const float* w1  = (const float*)d_in[7];
    const float* b1  = (const float*)d_in[8];
    const float* w2  = (const float*)d_in[9];
    const float* b2  = (const float*)d_in[10];
    float* out = (float*)d_out;

    char* ws = (char*)d_ws;
    float*  pooled  = (float*)(ws + 0);          // 32 KB
    float*  logits  = (float*)(ws + 32768);      // 64 KB
    int*    kval    = (int*)  (ws + 98304);      // 256 B slot
    int*    sel_idx = (int*)  (ws + 98560);      // 64 KB
    float*  sel_w   = (float*)(ws + 164096);     // 64 KB
    int*    sel_cnt = (int*)  (ws + 229632);     // 256 B slot
    __bf16* Hbuf    = (__bf16*)(ws + 229888);    // 32 MB (CHUNK_ x DFF_)
    __bf16* w1t     = (__bf16*)(ws + 229888 + 33554432);              // 32 MB [DFF_,D_]
    __bf16* w2t     = (__bf16*)(ws + 229888 + 2ull * 33554432);       // 32 MB [D_,DFF_]

    (void)hipMemsetAsync(sel_cnt, 0, sizeof(int) * B_, stream);
    pool_kernel<<<(B_ * D_) / 256, 256, 0, stream>>>(hidden, pooled);
    logits_kernel<<<(B_ * S_) / 8, 256, 0, stream>>>(hidden, rw, rb, logits);
    comp_kernel<<<1, 256, 0, stream>>>(pooled, cw1, cb1, cw2, cb2, kval);
    topk_kernel<<<B_ * (S_ / 256), 256, 0, stream>>>(logits, kval, sel_idx, sel_w, sel_cnt);
    (void)hipMemcpyAsync(out, hidden, (size_t)B_ * S_ * D_ * sizeof(float),
                         hipMemcpyDeviceToDevice, stream);

    // pre-transpose + convert weights to bf16 (L2-resident for the GEMMs)
    dim3 tg1(DFF_ / 32, D_ / 32);
    transpose_bf16_kernel<<<tg1, 256, 0, stream>>>(w1, w1t, D_, DFF_);
    dim3 tg2(D_ / 32, DFF_ / 32);
    transpose_bf16_kernel<<<tg2, 256, 0, stream>>>(w2, w2t, DFF_, D_);

    dim3 g1(DFF_ / 128, CHUNK_ / 64);
    dim3 g2(D_ / 128, CHUNK_ / 64);
    for (int b = 0; b < B_; ++b) {
        for (int c = 0; c < S_ / CHUNK_; ++c) {
            int row0 = c * CHUNK_;
            ffn1_kernel<<<g1, 256, 0, stream>>>(hidden, sel_idx + b * S_, kval,
                                                w1t, b1, Hbuf, b, row0);
            ffn2_kernel<<<g2, 256, 0, stream>>>(Hbuf, w2t, b2, sel_idx + b * S_,
                                                sel_w + b * S_, kval, out, b, row0);
        }
    }
}
